// Seq2Seq_33861522161913
// MI455X (gfx1250) — matrix operand order repaired and checked by reading, not compiled, not run on hardware
//
#include <hip/hip_runtime.h>

// ---------------------------------------------------------------------------
// Seq2seq LSTM (enc 8192 -> dec 4096 steps), HIDDEN=128, scalar in/out.
// Serial recurrence => single workgroup (8 x wave32). Per-step critical path:
//   8x ds_load_b128 (h fragment, all issued up-front; staggered dscnt waits)
//   -> 16x v_wmma_f32_16x16x32_f16 (4 independent gate chains, kc-outer)
//   -> in-register gate activations (wave w lane n owns hidden unit 16w+n;
//      sigmoid/tanh via v_exp_f32 + v_rcp_f32, no IEEE division)
//   -> 1x ds_store_b16 (new h) -> ONE barrier.
// B = Whh^T lives in 128 VGPRs/wave for the whole phase (f32->f16 from global
// once per phase). h and decoder fc-partials are parity double-buffered in LDS
// (576 B total) so a single barrier per step is race-free.
// ---------------------------------------------------------------------------

#define HID 128

typedef __attribute__((ext_vector_type(16))) _Float16 v16h;
typedef __attribute__((ext_vector_type(8)))  _Float16 v8h;
typedef __attribute__((ext_vector_type(8)))  float    v8f;

#define OFF_HF   0      // 2 x 128 f16 (double-buffered h)      : 512 B
#define OFF_FCP  512    // 2 x 8 f32 (double-buffered fc parts) :  64 B
#define SMEM_BYTES 576

#define LOG2E_F     1.442695041f
#define TWO_LOG2E_F 2.885390082f

struct BReg { v16h t[4][4]; };   // [gate q (i,f,g,o)][kchunk]

// sigmoid via v_exp_f32 + v_rcp_f32 (no IEEE division on the critical path)
__device__ __forceinline__ float sigm(float x) {
    return __builtin_amdgcn_rcpf(1.0f + __builtin_amdgcn_exp2f(-LOG2E_F * x));
}
// tanh(x) = 1 - 2/(exp(2x)+1), clamped so exp never overflows
__device__ __forceinline__ float tanh_f(float x) {
    x = fminf(fmaxf(x, -15.0f), 15.0f);
    float t = __builtin_amdgcn_rcpf(__builtin_amdgcn_exp2f(TWO_LOG2E_F * x) + 1.0f);
    return 1.0f - 2.0f * t;
}

// Register-resident B fragments (Whh^T tiles) from global f32 Whh.
// Wave w, gate q -> tile (w + 8q): rows 128q + 16w + [0,16).
// B element e of lane (half,n): K = kc*32 + 16*half + e, row = tile*16 + n.
__device__ __forceinline__ void load_breg(BReg& bf, const float* __restrict__ Whh,
                                          int tid) {
    const int lane = tid & 31, wave = tid >> 5;
    const int half = lane >> 4, n = lane & 15;
#pragma unroll
    for (int q = 0; q < 4; ++q) {
        const int row = (wave + 8 * q) * 16 + n;
#pragma unroll
        for (int kc = 0; kc < 4; ++kc) {
            const float* p = Whh + row * HID + kc * 32 + 16 * half;
            v16h b;
#pragma unroll
            for (int e = 0; e < 16; ++e) b[e] = (_Float16)p[e];
            bf.t[q][kc] = b;
        }
    }
}

// One LSTM step; hr = previous h (f16), hw = new h (other parity buffer).
template <bool DEC>
__device__ __forceinline__ void step(const BReg& bf, float x,
                                     const _Float16* __restrict__ hr,
                                     _Float16* __restrict__ hw,
                                     float* __restrict__ fcp_w,  // write-side partials
                                     float fcw, float& c,
                                     const float (&b4)[4], const float (&w4)[4],
                                     int tid) {
    const int lane = tid & 31, wave = tid >> 5;
    const int half = lane >> 4, n = lane & 15;

    // A fragment: h replicated over the 16 M rows, per 32-wide K chunk.
    // Issue ALL 8 ds_load_b128 before any WMMA so dscnt waits stagger.
    v16h a[4];
#pragma unroll
    for (int kc = 0; kc < 4; ++kc) {
        const int base = kc * 32 + 16 * half;
        v8h lo = *(const v8h*)(hr + base);
        v8h hi = *(const v8h*)(hr + base + 8);
        a[kc] = __builtin_shufflevector(lo, hi, 0, 1, 2, 3, 4, 5, 6, 7,
                                        8, 9, 10, 11, 12, 13, 14, 15);
    }
    __builtin_amdgcn_sched_barrier(0);   // keep loads above the WMMA block

    // 4 independent accumulation chains (one per gate) -> no WMMA RAW stalls.
    v8f acc[4] = {v8f{}, v8f{}, v8f{}, v8f{}};
#pragma unroll
    for (int kc = 0; kc < 4; ++kc)
#pragma unroll
        for (int q = 0; q < 4; ++q)
            acc[q] = __builtin_amdgcn_wmma_f32_16x16x32_f16(
                false, a[kc], false, bf.t[q][kc], (short)0, acc[q], false, false);

    // Lane n (0-15) of wave w now holds gates i,f,g,o of hidden unit 16w+n
    // in acc[0..3][0] (D rows identical; VGPR0 lanes 0-15 = (M=0, N=lane)).
    if (lane < 16) {
        float gi = acc[0][0] + b4[0] + w4[0] * x;
        float gf = acc[1][0] + b4[1] + w4[1] * x;
        float gg = acc[2][0] + b4[2] + w4[2] * x;
        float go = acc[3][0] + b4[3] + w4[3] * x;
        c = sigm(gf) * c + sigm(gi) * tanh_f(gg);
        float h = sigm(go) * tanh_f(c);
        hw[16 * wave + n] = (_Float16)h;
        if (DEC) {
            // per-wave fc partial: sum over this wave's 16 hidden units
            float p = fcw * h;
            p += __shfl_xor(p, 8, 32);
            p += __shfl_xor(p, 4, 32);
            p += __shfl_xor(p, 2, 32);
            p += __shfl_xor(p, 1, 32);
            if (n == 0) fcp_w[wave] = p;
        }
    }
    __syncthreads();
}

__device__ __forceinline__ float sum8(const float* __restrict__ p) {
    float4 s0 = *(const float4*)(p);
    float4 s1 = *(const float4*)(p + 4);
    return ((s0.x + s0.y) + (s0.z + s0.w)) + ((s1.x + s1.y) + (s1.z + s1.w));
}

__global__ void __launch_bounds__(256)
seq2seq_kernel(const float* __restrict__ input_seq,
               const float* __restrict__ enc_Wih, const float* __restrict__ enc_Whh,
               const float* __restrict__ enc_bih, const float* __restrict__ enc_bhh,
               const float* __restrict__ dec_Wih, const float* __restrict__ dec_Whh,
               const float* __restrict__ dec_bih, const float* __restrict__ dec_bhh,
               const float* __restrict__ fc_W,    const float* __restrict__ fc_b,
               float* __restrict__ out, int encT, int decT) {
    extern __shared__ char smem[];
    _Float16* hf     = (_Float16*)(smem + OFF_HF);   // [2][128]
    float*    fcpart = (float*)(smem + OFF_FCP);     // [2][8]

    const int tid = threadIdx.x;
    const int ju  = ((tid >> 5) << 4) | (tid & 15);  // hidden unit of act lane

    // init: zero both h buffers (256 halfs == 256 threads) and fc partials
    hf[tid] = (_Float16)0.0f;
    if (tid < 16) fcpart[tid] = 0.0f;
    float c = 0.0f;                                  // cell state of unit ju
    __syncthreads();

    BReg bf;
    float b4[4], w4[4];
    int rp = 0;                                      // parity: h_prev lives in hf[rp*128]

    // ================= encoder =================
    load_breg(bf, enc_Whh, tid);
#pragma unroll
    for (int g = 0; g < 4; ++g) {
        b4[g] = enc_bih[g * HID + ju] + enc_bhh[g * HID + ju];
        w4[g] = enc_Wih[g * HID + ju];
    }
#pragma unroll 1
    for (int t = 0; t < encT; ++t) {
        float x = input_seq[t];                      // uniform scalar load
        step<false>(bf, x, hf + rp * HID, hf + (rp ^ 1) * HID,
                    nullptr, 0.0f, c, b4, w4, tid);
        rp ^= 1;
    }

    // ================= decoder =================
    load_breg(bf, dec_Whh, tid);                     // swap register B to decoder Whh
#pragma unroll
    for (int g = 0; g < 4; ++g) {
        b4[g] = dec_bih[g * HID + ju] + dec_bhh[g * HID + ju];
        w4[g] = dec_Wih[g * HID + ju];
    }
    const float fcw = fc_W[ju];
    const float fcb = fc_b[0];

#pragma unroll 1
    for (int t = 0; t < decT; ++t) {
        // y_{t-1} = sum of previous step's per-wave fc partials (+ fc_b);
        // it is both this step's x input and out[t-1].
        float x = 0.0f;
        if (t > 0) {
            x = sum8(fcpart + rp * 8) + fcb;
            if (tid == 0) out[t - 1] = x;
        }
        step<true>(bf, x, hf + rp * HID, hf + (rp ^ 1) * HID,
                   fcpart + (rp ^ 1) * 8, fcw, c, b4, w4, tid);
        rp ^= 1;
    }
    // final prediction (partials written by the last step, after its barrier)
    if (tid == 0) out[decT - 1] = sum8(fcpart + rp * 8) + fcb;
}

extern "C" void kernel_launch(void* const* d_in, const int* in_sizes, int n_in,
                              void* d_out, int out_size, void* d_ws, size_t ws_size,
                              hipStream_t stream) {
    (void)n_in; (void)d_ws; (void)ws_size;
    const float* input_seq = (const float*)d_in[0];
    const float* enc_Wih   = (const float*)d_in[1];
    const float* enc_Whh   = (const float*)d_in[2];
    const float* enc_bih   = (const float*)d_in[3];
    const float* enc_bhh   = (const float*)d_in[4];
    const float* dec_Wih   = (const float*)d_in[5];
    const float* dec_Whh   = (const float*)d_in[6];
    const float* dec_bih   = (const float*)d_in[7];
    const float* dec_bhh   = (const float*)d_in[8];
    const float* fc_W      = (const float*)d_in[9];
    const float* fc_b      = (const float*)d_in[10];
    float*       out       = (float*)d_out;

    const int encT = in_sizes[0];   // 8192
    const int decT = out_size;      // 4096

    seq2seq_kernel<<<dim3(1), dim3(256), SMEM_BYTES, stream>>>(
        input_seq, enc_Wih, enc_Whh, enc_bih, enc_bhh,
        dec_Wih, dec_Whh, dec_bih, dec_bhh, fc_W, fc_b,
        out, encT, decT);
}
